// ManualNMRNN_53996328845819
// MI455X (gfx1250) — matrix operand order repaired and checked by reading, not compiled, not run on hardware
//
#include <hip/hip_runtime.h>
#include <math.h>

// ---------------------------------------------------------------------------
// ManualNMRNN forward for MI455X (gfx1250), wave32 + WMMA bf16 + TDM.
// B=128, T=512, I=256, H=512, Z=128, K=H=512.
//
// Pipeline:
//  1) one-time: inputs f32->bf16; all weights -> bf16 [N][K] transposed
//  2) batched GEMMs (M=B*T=65536) with TENSOR_LOAD_TO_LDS double buffering:
//     A_z = x@W_iz + b_z (ws), preH = x@W_ih + b_h (written into d_out)
//  3) whole z-recurrence in ONE persistent workgroup (state+W_zz in LDS)
//  4) per-step h kernel: s = sigmoid(z@W_zk+b_k); h = tanh((h@W_hh)*s + preH)
//  5) finalize: h_T / z_T tails
// Workspace use: ~82 MiB.
// ---------------------------------------------------------------------------

#define B_ 128
#define T_ 512
#define I_ 256
#define H_ 512
#define Z_ 128

typedef __attribute__((ext_vector_type(16))) __bf16 v16bf;
typedef __attribute__((ext_vector_type(8)))  float  v8f;
typedef __attribute__((ext_vector_type(4)))  unsigned int v4u;
typedef __attribute__((ext_vector_type(8)))  int v8i_;
typedef __attribute__((ext_vector_type(4)))  int v4i_;

struct U4 { unsigned int x, y, z, w; };
union FragB { v16bf v; U4 q[2]; unsigned short s[16]; };

__device__ __forceinline__ unsigned short f2bf(float f) {
  unsigned int u = __float_as_uint(f);
  u += 0x7FFFu + ((u >> 16) & 1u);          // round-to-nearest-even
  return (unsigned short)(u >> 16);
}

__device__ __forceinline__ v8f wmma_bf16(const FragB& a, const FragB& b, v8f c) {
  return __builtin_amdgcn_wmma_f32_16x16x32_bf16(
      false, a.v, false, b.v, (short)0, c, false, false);
}

// ---------------------------------------------------------------------------
// TDM: async 2D bf16 tile load, global -> LDS, with LDS padding of 8 dwords
// after every 16 dwords (reproduces the 48-element padded LDS row stride).
// D# packing per CDNA5 ISA ch.8 (group0: count/lds/global/type, group1:
// data_size|pad fields, dims, tile dims, row stride). Tensor dims are set
// huge so no OOB clipping ever triggers (tiles are always in-bounds).
// ---------------------------------------------------------------------------
#if defined(__has_include)
#if __has_include(<hip/amd_detail/amd_gfx1250_TDM.h>)
#define TDM_6ARG 1
#endif
#endif

__device__ __forceinline__ void tdm_load_2d_bf16(
    unsigned int lds_byte_off, const unsigned short* gptr,
    unsigned int tile_d0, unsigned int tile_d1, unsigned int row_stride_elems) {
  unsigned long long ga = (unsigned long long)(size_t)gptr;
  v4u g0;
  g0[0] = 1u;                                       // count=1, no gather
  g0[1] = lds_byte_off;                             // LDS dest (bytes)
  g0[2] = (unsigned int)ga;                         // global addr lo
  g0[3] = (unsigned int)(ga >> 32) | 0x80000000u;   // addr hi | type=2
  v8i_ g1;
  // data_size=1 (2B) | pad_enable | pad_interval=3 (16 dw) | pad_amount=7 (8 dw)
  g1[0] = (int)((1u << 16) | (1u << 20) | (3u << 22) | (7u << 25));
  g1[1] = (int)(0x7FFFu << 16);                     // tensor_dim0 lo16 (huge)
  g1[2] = (int)(0x0000u | (0xFFFFu << 16));         // dim0 hi16=0, dim1 lo16
  g1[3] = (int)(0x7FFFu | (tile_d0 << 16));         // dim1 hi16, tile_dim0
  g1[4] = (int)(tile_d1 & 0xFFFFu);                 // tile_dim1, tile_dim2=0
  g1[5] = (int)row_stride_elems;                    // tensor_dim0_stride lo32
  g1[6] = 0;                                        // stride hi, dim1_stride lo
  g1[7] = 0;
  v4i_ g2 = {};
  v4i_ g3 = {};
#ifdef TDM_6ARG
  v8i_ g4 = {};
  __builtin_amdgcn_tensor_load_to_lds(g0, g1, g2, g3, g4, 0);
#else
  __builtin_amdgcn_tensor_load_to_lds(g0, g1, g2, g3, 0);
#endif
}

// ---------------------------------------------------------------------------
// One-time data prep kernels
// ---------------------------------------------------------------------------
__global__ void convert_bf16_kernel(const float* __restrict__ X,
                                    unsigned short* __restrict__ Xb, int n) {
  int i = blockIdx.x * blockDim.x + threadIdx.x;
  if (i < n) Xb[i] = f2bf(X[i]);
}

// W[K][N] f32 -> Wt[N][K] bf16
__global__ void transpose_bf16_kernel(const float* __restrict__ W,
                                      unsigned short* __restrict__ Wt,
                                      int K, int N) {
  int idx = blockIdx.x * blockDim.x + threadIdx.x;
  if (idx >= K * N) return;
  int n = idx / K, k = idx - n * K;
  Wt[idx] = f2bf(W[(size_t)k * N + n]);
}

// ---------------------------------------------------------------------------
// Batched GEMM + bias:  Y[M][N] = Xbf[M][K] @ Wt[N][K]^T + bias[N]
// Block tile 128x64, K-step 32, 256 threads = 8 waves, 1 wave = 16 rows.
// TDM double-buffered LDS staging (wave 0 drives the DMA engine,
// s_wait_tensorcnt + workgroup barrier publish each tile); WMMA fragments
// are two b128 LDS loads each in the documented 16-bit lane layout.
// ---------------------------------------------------------------------------
#define P1_BM 128
#define P1_BN 64
#define P1_BK 32
#define P1_LD 48            // padded row stride (elements): 32 data + 16 pad

__global__ __launch_bounds__(256) void gemm_bias_tdm_kernel(
    const unsigned short* __restrict__ Xbf,   // [M][K] bf16
    const unsigned short* __restrict__ Wt,    // [N][K] bf16
    const float* __restrict__ bias, float* __restrict__ Y,
    int N, int K) {
  __shared__ unsigned short As[2][P1_BM * P1_LD];   // 2 x 12 KiB
  __shared__ unsigned short Bs[2][P1_BN * P1_LD];   // 2 x 6 KiB

  const int tid   = threadIdx.x;
  const int wid   = tid >> 5;
  const int lane  = tid & 31;
  const int lhalf = lane >> 4;
  const int lrow  = lane & 15;
  const int m0 = blockIdx.y * P1_BM;
  const int n0 = blockIdx.x * P1_BN;

  v8f acc[4];
  for (int j = 0; j < 4; ++j) acc[j] = {};

  // Prime buffer 0
  if (tid == 0) {
    tdm_load_2d_bf16((unsigned int)(size_t)&As[0][0],
                     Xbf + (size_t)m0 * K, P1_BK, P1_BM, K);
    tdm_load_2d_bf16((unsigned int)(size_t)&Bs[0][0],
                     Wt + (size_t)n0 * K, P1_BK, P1_BN, K);
    __builtin_amdgcn_s_wait_tensorcnt(0);
  }
  __syncthreads();

  int buf = 0;
  for (int k0 = 0; k0 < K; k0 += P1_BK) {
    // Kick DMA for next K-slice into the other buffer
    if (tid == 0 && (k0 + P1_BK) < K) {
      tdm_load_2d_bf16((unsigned int)(size_t)&As[buf ^ 1][0],
                       Xbf + (size_t)m0 * K + (k0 + P1_BK), P1_BK, P1_BM, K);
      tdm_load_2d_bf16((unsigned int)(size_t)&Bs[buf ^ 1][0],
                       Wt + (size_t)n0 * K + (k0 + P1_BK), P1_BK, P1_BN, K);
    }

    // Compute on current buffer
    FragB a;
    int arow = wid * 16 + lrow;
    int ab = lhalf * 8;
    a.q[0] = *(const U4*)&As[buf][arow * P1_LD + ab];
    a.q[1] = *(const U4*)&As[buf][arow * P1_LD + ab + 16];
    for (int j = 0; j < 4; ++j) {
      FragB b;
      int bcol = j * 16 + lrow;
      int kb = lhalf * 8;
      b.q[0] = *(const U4*)&Bs[buf][bcol * P1_LD + kb];
      b.q[1] = *(const U4*)&Bs[buf][bcol * P1_LD + kb + 16];
      acc[j] = wmma_bf16(a, b, acc[j]);
    }

    if (tid == 0) __builtin_amdgcn_s_wait_tensorcnt(0);
    __syncthreads();                        // next buffer ready + reads done
    buf ^= 1;
  }

  for (int j = 0; j < 4; ++j) {
    int colg = n0 + j * 16 + lrow;
    float bv = bias[colg];
    for (int v = 0; v < 8; ++v) {
      int rowg = m0 + wid * 16 + v + lhalf * 8;
      Y[(size_t)rowg * N + colg] = acc[j][v] + bv;
    }
  }
}

// ---------------------------------------------------------------------------
// Full z-recurrence in one persistent workgroup (1024 thr = 32 waves).
// z state (128x128 bf16, 32KB) and W_zz^T (32KB) live in LDS; each wave owns
// two 16x16 output tiles; 512 steps separated by __syncthreads().
// ---------------------------------------------------------------------------
__global__ __launch_bounds__(1024) void zchain_kernel(
    const unsigned short* __restrict__ Wzz_t,   // [n=128][k=128] bf16
    const float* __restrict__ A_z,              // (B*T) x Z : rows b*T+t
    unsigned short* __restrict__ z_all,         // T x B x Z bf16
    float* __restrict__ z_last) {               // B x Z f32
  __shared__ unsigned short zs[B_ * Z_];
  __shared__ unsigned short wz[Z_ * Z_];

  const int tid   = threadIdx.x;
  const int wid   = tid >> 5;
  const int lane  = tid & 31;
  const int lhalf = lane >> 4;
  const int lrow  = lane & 15;

  for (int i = tid; i < Z_ * Z_; i += 1024) wz[i] = Wzz_t[i];
  for (int i = tid; i < B_ * Z_; i += 1024) zs[i] = 0;    // z0 = 0
  __syncthreads();

  for (int t = 0; t < T_; ++t) {
    float res[2][8];
    for (int p = 0; p < 2; ++p) {
      int tileId = wid * 2 + p;            // 0..63 (8x8 tiles)
      int tm = tileId >> 3, tn = tileId & 7;
      int arow = tm * 16 + lrow;
      int bcol = tn * 16 + lrow;
      v8f acc = {};
      for (int kk = 0; kk < Z_; kk += 32) {
        FragB a, b;
        int ab = kk + lhalf * 8;
        a.q[0] = *(const U4*)&zs[arow * Z_ + ab];
        a.q[1] = *(const U4*)&zs[arow * Z_ + ab + 16];
        int bb = kk + lhalf * 8;
        b.q[0] = *(const U4*)&wz[bcol * Z_ + bb];
        b.q[1] = *(const U4*)&wz[bcol * Z_ + bb + 16];
        acc = wmma_bf16(a, b, acc);
      }
      for (int v = 0; v < 8; ++v) {
        int rowb = tm * 16 + v + lhalf * 8;
        res[p][v] = tanhf(acc[v] + A_z[((size_t)rowb * T_ + t) * Z_ + bcol]);
      }
    }
    __syncthreads();                       // all reads of zs done
    for (int p = 0; p < 2; ++p) {
      int tileId = wid * 2 + p;
      int tm = tileId >> 3, tn = tileId & 7;
      int col = tn * 16 + lrow;
      for (int v = 0; v < 8; ++v) {
        int rowb = tm * 16 + v + lhalf * 8;
        unsigned short zb = f2bf(res[p][v]);
        zs[rowb * Z_ + col] = zb;
        z_all[((size_t)t * B_ + rowb) * Z_ + col] = zb;
        if (t == T_ - 1) z_last[rowb * Z_ + col] = res[p][v];
      }
    }
    __syncthreads();                       // zs updated for next step
  }
}

// ---------------------------------------------------------------------------
// One h-step: 256 output tiles (8x32), 32 blocks x 8 waves, 1 tile/wave.
// acc_s over K=128 (z_t @ W_zk), acc_h over K=512 (h_{t-1} @ W_hh);
// h = tanh(acc_h * sigmoid(acc_s + b_k) + preH), stored into d_out in place.
// All weight/state fragments are b128 global loads from L2-resident bf16.
// ---------------------------------------------------------------------------
__global__ __launch_bounds__(256) void hstep_kernel(
    const unsigned short* __restrict__ h_prev,  // B x H bf16
    const unsigned short* __restrict__ z_cur,   // B x Z bf16 (z_all + t*B*Z)
    const unsigned short* __restrict__ Whh_t,   // [n=512][k=512] bf16
    const unsigned short* __restrict__ Wzk_t,   // [n=512][k=128] bf16
    const float* __restrict__ bias_k,
    float* __restrict__ hidden,                 // d_out: preact in, h out
    unsigned short* __restrict__ h_next,        // B x H bf16
    int t) {
  const int tid   = threadIdx.x;
  const int wid   = tid >> 5;
  const int lane  = tid & 31;
  const int lhalf = lane >> 4;
  const int lrow  = lane & 15;
  const int tileId = blockIdx.x * 8 + wid;   // 0..255
  const int tm = tileId >> 5;                // 0..7
  const int tn = tileId & 31;                // 0..31
  const int arow = tm * 16 + lrow;
  const int bcol = tn * 16 + lrow;

  v8f acc_s = {}, acc_h = {};
  for (int kk = 0; kk < Z_; kk += 32) {      // z_t @ W_zk
    FragB a, b;
    int ab = kk + lhalf * 8;
    a.q[0] = *(const U4*)&z_cur[arow * Z_ + ab];
    a.q[1] = *(const U4*)&z_cur[arow * Z_ + ab + 16];
    int bb = kk + lhalf * 8;
    b.q[0] = *(const U4*)&Wzk_t[(size_t)bcol * Z_ + bb];
    b.q[1] = *(const U4*)&Wzk_t[(size_t)bcol * Z_ + bb + 16];
    acc_s = wmma_bf16(a, b, acc_s);
  }
  for (int kk = 0; kk < H_; kk += 32) {      // h_{t-1} @ W_hh
    FragB a, b;
    int ab = kk + lhalf * 8;
    a.q[0] = *(const U4*)&h_prev[arow * H_ + ab];
    a.q[1] = *(const U4*)&h_prev[arow * H_ + ab + 16];
    int bb = kk + lhalf * 8;
    b.q[0] = *(const U4*)&Whh_t[(size_t)bcol * H_ + bb];
    b.q[1] = *(const U4*)&Whh_t[(size_t)bcol * H_ + bb + 16];
    acc_h = wmma_bf16(a, b, acc_h);
  }

  float bk = bias_k[bcol];
  for (int v = 0; v < 8; ++v) {
    int rowb = tm * 16 + v + lhalf * 8;
    float s = 1.0f / (1.0f + __expf(-(acc_s[v] + bk)));
    size_t idx = ((size_t)rowb * T_ + t) * H_ + bcol;
    float h = tanhf(acc_h[v] * s + hidden[idx]);
    hidden[idx] = h;
    h_next[rowb * H_ + bcol] = f2bf(h);
  }
}

// ---------------------------------------------------------------------------
__global__ void init_h_kernel(unsigned short* __restrict__ hbufs) {
  int i = blockIdx.x * blockDim.x + threadIdx.x;
  if (i < 2 * B_ * H_) hbufs[i] = 0;
}

__global__ void finalize_kernel(const float* __restrict__ hidden,
                                const float* __restrict__ z_last,
                                float* __restrict__ out_tail) {
  int i = blockIdx.x * blockDim.x + threadIdx.x;
  if (i < B_ * H_) {
    int b = i >> 9, h = i & (H_ - 1);
    out_tail[i] = hidden[((size_t)b * T_ + (T_ - 1)) * H_ + h];
  } else if (i < B_ * H_ + B_ * Z_) {
    out_tail[i] = z_last[i - B_ * H_];
  }
}

// ---------------------------------------------------------------------------
extern "C" void kernel_launch(void* const* d_in, const int* in_sizes, int n_in,
                              void* d_out, int out_size, void* d_ws, size_t ws_size,
                              hipStream_t stream) {
  (void)in_sizes; (void)n_in; (void)out_size; (void)ws_size;

  const float* inputs = (const float*)d_in[0];
  const float* W_ih   = (const float*)d_in[1];
  const float* W_hh   = (const float*)d_in[2];
  const float* W_iz   = (const float*)d_in[3];
  const float* W_zz   = (const float*)d_in[4];
  const float* W_zk   = (const float*)d_in[5];
  const float* bias_z = (const float*)d_in[6];
  const float* bias_h = (const float*)d_in[7];
  const float* bias_k = (const float*)d_in[8];

  float* hidden   = (float*)d_out;                       // B*T*H
  float* out_tail = hidden + (size_t)B_ * T_ * H_;       // h_T then z_T

  char* ws = (char*)d_ws;
  size_t off = 0;
  float* A_z = (float*)(ws + off);                     off += (size_t)B_ * T_ * I_;       // dummy spacing avoided below
  // NOTE: recompute layout explicitly:
  off = 0;
  A_z = (float*)(ws + off);                            off += (size_t)B_ * T_ * Z_ * 4;   // 32 MiB
  unsigned short* x_bf  = (unsigned short*)(ws + off); off += (size_t)B_ * T_ * I_ * 2;   // 32 MiB
  unsigned short* Whh_t = (unsigned short*)(ws + off); off += (size_t)H_ * H_ * 2;
  unsigned short* Wzk_t = (unsigned short*)(ws + off); off += (size_t)H_ * Z_ * 2;
  unsigned short* Wzz_t = (unsigned short*)(ws + off); off += (size_t)Z_ * Z_ * 2;
  unsigned short* Wih_t = (unsigned short*)(ws + off); off += (size_t)H_ * I_ * 2;
  unsigned short* Wiz_t = (unsigned short*)(ws + off); off += (size_t)Z_ * I_ * 2;
  unsigned short* z_all = (unsigned short*)(ws + off); off += (size_t)T_ * B_ * Z_ * 2;   // 16 MiB
  unsigned short* hbufs = (unsigned short*)(ws + off); off += (size_t)2 * B_ * H_ * 2;
  float* z_last = (float*)(ws + off);                  off += (size_t)B_ * Z_ * 4;

  // 1) one-time conversions / transposes to bf16
  convert_bf16_kernel<<<(B_ * T_ * I_ + 255) / 256, 256, 0, stream>>>(
      inputs, x_bf, B_ * T_ * I_);
  transpose_bf16_kernel<<<(H_ * H_ + 255) / 256, 256, 0, stream>>>(W_hh, Whh_t, H_, H_);
  transpose_bf16_kernel<<<(Z_ * H_ + 255) / 256, 256, 0, stream>>>(W_zk, Wzk_t, Z_, H_);
  transpose_bf16_kernel<<<(Z_ * Z_ + 255) / 256, 256, 0, stream>>>(W_zz, Wzz_t, Z_, Z_);
  transpose_bf16_kernel<<<(I_ * H_ + 255) / 256, 256, 0, stream>>>(W_ih, Wih_t, I_, H_);
  transpose_bf16_kernel<<<(I_ * Z_ + 255) / 256, 256, 0, stream>>>(W_iz, Wiz_t, I_, Z_);
  init_h_kernel<<<(2 * B_ * H_ + 255) / 256, 256, 0, stream>>>(hbufs);

  // 2) batched input GEMMs (M = B*T = 65536), TDM double-buffered
  {
    dim3 g(Z_ / P1_BN, (B_ * T_) / P1_BM);
    gemm_bias_tdm_kernel<<<g, 256, 0, stream>>>(x_bf, Wiz_t, bias_z, A_z, Z_, I_);
  }
  {
    dim3 g(H_ / P1_BN, (B_ * T_) / P1_BM);
    gemm_bias_tdm_kernel<<<g, 256, 0, stream>>>(x_bf, Wih_t, bias_h, hidden, H_, I_);
  }

  // 3) entire z-recurrence: one persistent workgroup
  zchain_kernel<<<1, 1024, 0, stream>>>(Wzz_t, A_z, z_all, z_last);

  // 4) h-recurrence: one kernel per step, ping-pong bf16 h state
  for (int t = 0; t < T_; ++t) {
    const unsigned short* hp = hbufs + (size_t)(t & 1) * B_ * H_;
    unsigned short*       hn = hbufs + (size_t)((t + 1) & 1) * B_ * H_;
    const unsigned short* zc = z_all + (size_t)t * B_ * Z_;
    hstep_kernel<<<32, 256, 0, stream>>>(hp, zc, Whh_t, Wzk_t, bias_k, hidden, hn, t);
  }

  // 5) tails: h_T, z_T
  finalize_kernel<<<(B_ * H_ + B_ * Z_ + 255) / 256, 256, 0, stream>>>(hidden, z_last, out_tail);
}
